// FFTMemory_39204461478043
// MI455X (gfx1250) — compile-verified, matrix operand order checked by reading, not run on hardware
//
#include <hip/hip_runtime.h>
#include <hip/hip_bf16.h>
#include <math.h>

// ---------------- types ----------------
typedef __attribute__((ext_vector_type(16))) __bf16 v16bf;
typedef __attribute__((ext_vector_type(8)))  __bf16 v8bf;
typedef __attribute__((ext_vector_type(8)))  float  v8f;
typedef __attribute__((ext_vector_type(4)))  float  v4f;

union V16U { v16bf v; v8bf h[2]; };

#define Bq    2048
#define Cdim  1024
#define HW    49
#define Mkeys 65536
#define TOPK  32
#define BTILE 16       // batch rows per workgroup
#define NTILE 256      // sim columns per iteration (8 waves x 32)
#define SPLIT 4        // split of M across blockIdx.y
#define MCHUNK (Mkeys / SPLIT)
#define NEG_BIG (-1e30f)
#define EPSN 1e-12f

// ---------------- kernel 1: L2-normalize keys -> bf16 ----------------
// one wave (32 lanes) per key row; 8 rows per 256-thread block
__global__ void knorm_kernel(const float* __restrict__ keys,
                             __bf16* __restrict__ kn) {
    const int row  = blockIdx.x * 8 + (threadIdx.x >> 5);
    const int lane = threadIdx.x & 31;
    const float* src = keys + (size_t)row * Cdim;

    float vals[32];
    float ss = 0.f;
#pragma unroll
    for (int i = 0; i < 32; ++i) {
        float v = src[lane + i * 32];   // coalesced
        vals[i] = v;
        ss += v * v;
    }
#pragma unroll
    for (int off = 16; off > 0; off >>= 1) ss += __shfl_xor(ss, off, 32);
    const float scale = 1.f / fmaxf(sqrtf(ss), EPSN);

    __bf16* dst = kn + (size_t)row * Cdim;
#pragma unroll
    for (int i = 0; i < 32; ++i) dst[lane + i * 32] = (__bf16)(vals[i] * scale);
}

// ---------------- kernel 2: spatial mean-pool + L2-normalize q -> bf16 ----------------
// one 256-thread block per batch row; thread t owns channels t, t+256, t+512, t+768
__global__ void qnorm_kernel(const float* __restrict__ q,
                             __bf16* __restrict__ qn) {
    const int b = blockIdx.x;
    const int t = threadIdx.x;
    float m[4];
    float ss = 0.f;
#pragma unroll
    for (int j = 0; j < 4; ++j) {
        const int c = t + j * 256;
        const float* p = q + ((size_t)b * Cdim + c) * HW;
        float s = 0.f;
#pragma unroll
        for (int i = 0; i < HW; ++i) s += p[i];
        m[j] = s * (1.f / (float)HW);
        ss += m[j] * m[j];
    }
    // block reduction of ss
    __shared__ float red[8];
#pragma unroll
    for (int off = 16; off > 0; off >>= 1) ss += __shfl_xor(ss, off, 32);
    if ((t & 31) == 0) red[t >> 5] = ss;
    __syncthreads();
    float tot = 0.f;
#pragma unroll
    for (int w = 0; w < 8; ++w) tot += red[w];
    const float scale = 1.f / fmaxf(sqrtf(tot), EPSN);

    __bf16* dst = qn + (size_t)b * Cdim;
#pragma unroll
    for (int j = 0; j < 4; ++j) dst[t + j * 256] = (__bf16)(m[j] * scale);
}

// ---------------- top-k insertion helper ----------------
__device__ __forceinline__ void topk_insert(float v, int idx,
                                            float* vals, int* idxs,
                                            float& vmin, int& smin) {
    if (v > vmin) {
        vals[smin] = v;
        idxs[smin] = idx;
        vmin = vals[0]; smin = 0;
#pragma unroll
        for (int t = 1; t < TOPK; ++t) {
            float x = vals[t];
            if (x < vmin) { vmin = x; smin = t; }
        }
    }
}

// ---------------- kernel 3: fused bf16 WMMA GEMM + per-row top-32 (split over M) --------
// grid (B/16, SPLIT), 256 threads (8 waves). Wave w computes a 16x32 sim tile
// (two WMMA accumulators sharing one A fragment); threads 0..15 each maintain the
// top-32 list for one batch row in LDS with a vectorized pre-filtered scan.
__global__ void __launch_bounds__(256)
sim_topk_kernel(const __bf16* __restrict__ qn,
                const __bf16* __restrict__ kn,
                float* __restrict__ partVal,
                int*   __restrict__ partIdx) {
    const int tid    = threadIdx.x;
    const int wave   = tid >> 5;
    const int lane   = tid & 31;
    const int half   = lane >> 4;          // 0: K {0..7,16..23}, 1: K {8..15,24..31}
    const int l16    = lane & 15;
    const int b_base = blockIdx.x * BTILE;
    const int split  = blockIdx.y;

    __shared__ float simLds[BTILE][NTILE + 4];   // row stride 260 floats (16B aligned)
    __shared__ float topVal[BTILE][TOPK];
    __shared__ int   topIdx[BTILE][TOPK];

    if (tid < BTILE) {
#pragma unroll
        for (int t = 0; t < TOPK; ++t) { topVal[tid][t] = NEG_BIG; topIdx[tid][t] = 0; }
    }
    float vmin = NEG_BIG;
    int   smin = 0;
    __syncthreads();

    const int koff = half ? 8 : 0;                       // first 8-element K chunk
    const __bf16* aRow = qn + (size_t)(b_base + l16) * Cdim + koff;

    for (int it = 0; it < MCHUNK / NTILE; ++it) {
        const int m0 = split * MCHUNK + it * NTILE;
        const __bf16* bRow0 = kn + (size_t)(m0 + wave * 32 + l16) * Cdim + koff;
        const __bf16* bRow1 = bRow0 + (size_t)16 * Cdim;

        v8f acc0 = {};
        v8f acc1 = {};
#pragma unroll 4
        for (int kk = 0; kk < Cdim; kk += 32) {
            V16U a, b0, b1;
            a.h[0]  = *(const v8bf*)(aRow  + kk);
            a.h[1]  = *(const v8bf*)(aRow  + kk + 16);
            b0.h[0] = *(const v8bf*)(bRow0 + kk);
            b0.h[1] = *(const v8bf*)(bRow0 + kk + 16);
            b1.h[0] = *(const v8bf*)(bRow1 + kk);
            b1.h[1] = *(const v8bf*)(bRow1 + kk + 16);
            acc0 = __builtin_amdgcn_wmma_f32_16x16x32_bf16(
                       false, a.v, false, b0.v, (short)0, acc0, false, false);
            acc1 = __builtin_amdgcn_wmma_f32_16x16x32_bf16(
                       false, a.v, false, b1.v, (short)0, acc1, false, false);
        }
        // scatter C tiles to LDS: VGPR i, lane<16 -> (M=i,N=l16); lane>=16 -> (M=i+8,N=l16)
        const int scol = wave * 32 + l16;
        const int rb   = half * 8;
#pragma unroll
        for (int i = 0; i < 8; ++i) {
            simLds[rb + i][scol]      = acc0[i];
            simLds[rb + i][scol + 16] = acc1[i];
        }
        __syncthreads();

        // per-row top-32 maintenance (thread r owns row r), float4 + 4-way max pre-filter
        if (tid < BTILE) {
            const v4f* rowp = (const v4f*)(&simLds[tid][0]);
            float* tv = topVal[tid];
            int*   ti = topIdx[tid];
#pragma unroll 8
            for (int j4 = 0; j4 < NTILE / 4; ++j4) {
                const v4f c = rowp[j4];
                const float mx = fmaxf(fmaxf(c[0], c[1]), fmaxf(c[2], c[3]));
                if (mx > vmin) {
#pragma unroll
                    for (int e = 0; e < 4; ++e)
                        topk_insert(c[e], m0 + j4 * 4 + e, tv, ti, vmin, smin);
                }
            }
        }
        __syncthreads();
    }

    // write this split's partial top-32 lists
    for (int e = tid; e < BTILE * TOPK; e += 256) {
        const int r = e >> 5, t = e & 31;
        const size_t o = ((size_t)(b_base + r) * SPLIT + split) * TOPK + t;
        partVal[o] = topVal[r][t];
        partIdx[o] = topIdx[r][t];
    }
}

// ---------------- kernel 4: merge split top-32s, softmax, weighted gather ---------------
// grid B/16, 256 threads. Threads 0..15 merge/softmax their row; then all 256 threads
// do the float4-coalesced weighted sum over the 32 selected value rows.
__global__ void __launch_bounds__(256)
merge_gather_kernel(const float* __restrict__ partVal,
                    const int*   __restrict__ partIdx,
                    const float* __restrict__ values,
                    float* __restrict__ out) {
    const int tid    = threadIdx.x;
    const int b_base = blockIdx.x * BTILE;

    __shared__ float tv[BTILE][TOPK];
    __shared__ int   ti[BTILE][TOPK];
    __shared__ float wt[BTILE][TOPK];

    if (tid < BTILE) {
#pragma unroll
        for (int t = 0; t < TOPK; ++t) { tv[tid][t] = NEG_BIG; ti[tid][t] = 0; }
        float vmin = NEG_BIG;
        int   smin = 0;
        const size_t base = (size_t)(b_base + tid) * SPLIT * TOPK;
        for (int j = 0; j < SPLIT * TOPK; ++j)
            topk_insert(partVal[base + j], partIdx[base + j], tv[tid], ti[tid], vmin, smin);

        // softmax over the 32 winners (order-invariant)
        float vmax = tv[tid][0];
#pragma unroll
        for (int t = 1; t < TOPK; ++t) vmax = fmaxf(vmax, tv[tid][t]);
        float sum = 0.f;
#pragma unroll
        for (int t = 0; t < TOPK; ++t) { float e = __expf(tv[tid][t] - vmax); wt[tid][t] = e; sum += e; }
        const float inv = 1.f / sum;
#pragma unroll
        for (int t = 0; t < TOPK; ++t) wt[tid][t] *= inv;
    }
    __syncthreads();

    const v4f* v4 = (const v4f*)values;      // row stride Cdim/4 = 256 float4
    v4f*       o4 = (v4f*)out;
    for (int r = 0; r < BTILE; ++r) {
        v4f a = {};
#pragma unroll 8
        for (int k = 0; k < TOPK; ++k) {
            const float w = wt[r][k];
            const v4f  vv = v4[(size_t)ti[r][k] * (Cdim / 4) + tid];
            a += w * vv;
        }
        o4[(size_t)(b_base + r) * (Cdim / 4) + tid] = a;
    }
}

// ---------------- launcher ----------------
extern "C" void kernel_launch(void* const* d_in, const int* in_sizes, int n_in,
                              void* d_out, int out_size, void* d_ws, size_t ws_size,
                              hipStream_t stream) {
    const float* q      = (const float*)d_in[0];   // [2048,1024,7,7]
    const float* keys   = (const float*)d_in[1];   // [65536,1024]
    const float* values = (const float*)d_in[2];   // [65536,1024]
    float* out = (float*)d_out;                    // [2048,1024,1,1]

    char* ws = (char*)d_ws;
    __bf16* kn      = (__bf16*)ws;                                   // 134 MB
    __bf16* qn      = (__bf16*)(ws + (size_t)Mkeys * Cdim * 2);      //   4 MB
    float*  partVal = (float*)(ws + (size_t)Mkeys * Cdim * 2
                                  + (size_t)Bq * Cdim * 2);          //   1 MB
    int*    partIdx = (int*)((char*)partVal
                                  + (size_t)Bq * SPLIT * TOPK * 4);  //   1 MB

    knorm_kernel<<<Mkeys / 8, 256, 0, stream>>>(keys, kn);
    qnorm_kernel<<<Bq, 256, 0, stream>>>(q, qn);
    sim_topk_kernel<<<dim3(Bq / BTILE, SPLIT), 256, 0, stream>>>(qn, kn, partVal, partIdx);
    merge_gather_kernel<<<Bq / BTILE, 256, 0, stream>>>(partVal, partIdx, values, out);
}
